// hyperRNN_86964497809429
// MI455X (gfx1250) — matrix operand order repaired and checked by reading, not compiled, not run on hardware
//
#include <hip/hip_runtime.h>
#include <hip/hip_bf16.h>

typedef __attribute__((ext_vector_type(16))) __bf16 v16bf;
typedef __attribute__((ext_vector_type(8)))  float  v8f;

#define EPSF     1e-15f
#define MAXNORM  (1.0f - 1e-5f)
#define KDIM     2048
#define NDIM     2048
#define BATCH    64
#define TSTEPS   256
#define KTILES   (KDIM / 32)    // 64
#define NTILES   (NDIM / 16)    // 128

// ---------------- CDNA5 async global->LDS helpers ----------------

__device__ __forceinline__ void async_cp16(const void* lds_generic, const void* gsrc) {
  // GLOBAL_LOAD_ASYNC_TO_LDS_B128: VDST = LDS byte address, VADDR = 64-bit global.
  unsigned laddr = (unsigned)(uintptr_t)lds_generic;
  asm volatile("global_load_async_to_lds_b128 %0, %1, off"
               : : "v"(laddr), "v"(gsrc) : "memory");
}

template <int N>
__device__ __forceinline__ void wait_asynccnt() {
#if __has_builtin(__builtin_amdgcn_s_wait_asynccnt)
  __builtin_amdgcn_s_wait_asynccnt(N);
#else
  asm volatile("s_wait_asynccnt %0" : : "n"(N));
#endif
}

// ---------------- generic helpers ----------------

__device__ __forceinline__ float block_reduce_sum(float v, float* red) {
  const int tid = threadIdx.x;
  red[tid] = v;
  __syncthreads();
  #pragma unroll
  for (int s = 128; s > 0; s >>= 1) {
    if (tid < s) red[tid] += red[tid + s];
    __syncthreads();
  }
  float r = red[0];
  __syncthreads();
  return r;
}

__device__ __forceinline__ float atanh_ref(float x) {
  x = fminf(x, 1.0f - EPSF);
  return 0.5f * (log1pf(x + EPSF) - log1pf(EPSF - x));
}

// device-scope grid barrier (16 resident WGs): release-arrive + acquire-spin
__device__ __forceinline__ void grid_sync(unsigned* sync, unsigned target) {
  __threadfence();
  __syncthreads();
  if (threadIdx.x == 0) {
    __hip_atomic_fetch_add(sync, 1u, __ATOMIC_RELEASE, __HIP_MEMORY_SCOPE_AGENT);
    while (__hip_atomic_load(sync, __ATOMIC_ACQUIRE, __HIP_MEMORY_SCOPE_AGENT) < target)
      __builtin_amdgcn_s_sleep(1);
  }
  __syncthreads();
}

// ---------------- operand pre-swizzle into WMMA fragment order ----------------
// B fragment (32K x 16N), wave32 layout: lane = (n&15) + 16*((k>>3)&1), pos = (k&7) + 8*(k>=16).
__global__ __launch_bounds__(256) void swizzle_weights(const float* __restrict__ W,
                                                       __bf16* __restrict__ Bsw) {
  const int frag = blockIdx.x * 8 + (threadIdx.x >> 5);   // kt*128 + nt
  const int lane = threadIdx.x & 31;
  const int kt = frag >> 7, nt = frag & 127;
  const int col   = nt * 16 + (lane & 15);
  const int kbase = kt * 32 + ((lane >> 4) ? 8 : 0);
  v16bf v;
  #pragma unroll
  for (int p = 0; p < 16; ++p)
    v[p] = (__bf16)W[(size_t)(kbase + p + (p >= 8 ? 8 : 0)) * NDIM + col];
  *(v16bf*)(Bsw + (size_t)frag * 512 + lane * 16) = v;
}

// A fragment (16M x 32K), wave32 layout:
//   lanes 0-15 (row rl): pos 0-7 <-> K 0-7,  pos 8-15 <-> K 16-23
//   lanes 16-31(row rl): pos 0-7 <-> K 8-15, pos 8-15 <-> K 24-31
// Panel: f = (mb*KTILES + kt)*4 + st ; each (mb,kt) block = 4 frags = 4KB contiguous.
__global__ __launch_bounds__(256) void swizzle_x(const float* __restrict__ X,
                                                 __bf16* __restrict__ Asw) {
  const int f    = blockIdx.x * 8 + (threadIdx.x >> 5);
  const int lane = threadIdx.x & 31;
  const int st = f & 3, kt = (f >> 2) & (KTILES - 1), mb = f >> 8;
  const int row = mb * 64 + st * 16 + (lane & 15);
  const int kb  = kt * 32 + ((lane >> 4) ? 8 : 0);
  v16bf v;
  #pragma unroll
  for (int j = 0; j < 8; ++j) {
    v[j]     = (__bf16)(X[(size_t)row * KDIM + kb + j] + EPSF);
    v[8 + j] = (__bf16)(X[(size_t)row * KDIM + kb + 16 + j] + EPSF);
  }
  *(v16bf*)(Asw + (size_t)f * 512 + lane * 16) = v;
}

// per-row norm of (x + EPS)
__global__ __launch_bounds__(256) void row_norm(const float* __restrict__ x,
                                                float* __restrict__ xn) {
  __shared__ float red[256];
  const long long base = (long long)blockIdx.x * KDIM;
  float s = 0.f;
  #pragma unroll
  for (int i = 0; i < 8; ++i) {
    float v = x[base + threadIdx.x + i * 256] + EPSF;
    s += v * v;
  }
  s = block_reduce_sum(s, red);
  if (threadIdx.x == 0) xn[blockIdx.x] = sqrtf(s);
}

// ---------------- shared WMMA GEMM panel ----------------
// Wave tile 64M x (16*NSUB)N; WG = 8 waves -> 64M x (128*NSUB)N.
// A block (4KB / K-step) async-copied to LDS, double buffered; B frags direct from L2.
template <int NSUB>
__device__ __forceinline__ void gemm_panel(const __bf16* __restrict__ Apanel,
                                           const __bf16* __restrict__ Bsw,
                                           float* __restrict__ C, int mbase,
                                           __bf16 (&As)[2][2048]) {
  const int tid  = threadIdx.x;
  const int wave = tid >> 5;
  const int lane = tid & 31;
  const int hs   = lane >> 4;
  const int rl   = lane & 15;
  const int nt0  = blockIdx.x * 8 * NSUB + wave * NSUB;

  v8f acc[4][NSUB] = {};

  async_cp16(&As[0][tid * 8], Apanel + tid * 8);

  for (int kt = 0; kt < KTILES; ++kt) {
    const int cur = kt & 1;
    if (kt + 1 < KTILES) {
      async_cp16(&As[cur ^ 1][tid * 8], Apanel + (size_t)(kt + 1) * 2048 + tid * 8);
      wait_asynccnt<1>();
    } else {
      wait_asynccnt<0>();
    }
    __syncthreads();

    if (kt + 8 < KTILES)
      __builtin_prefetch(Bsw + ((size_t)(kt + 8) * NTILES + nt0) * 512 + lane * 16, 0, 1);

    v16bf bfr[NSUB];
    #pragma unroll
    for (int j = 0; j < NSUB; ++j)
      bfr[j] = *(const v16bf*)(Bsw + ((size_t)kt * NTILES + nt0 + j) * 512 + lane * 16);

    #pragma unroll
    for (int st = 0; st < 4; ++st) {
      const v16bf afr = *(const v16bf*)(&As[cur][st * 512 + lane * 16]);
      #pragma unroll
      for (int j = 0; j < NSUB; ++j)
        acc[st][j] = __builtin_amdgcn_wmma_f32_16x16x32_bf16(false, afr, false, bfr[j],
                                                             (short)0, acc[st][j], false, false);
    }
    __syncthreads();
  }

  #pragma unroll
  for (int st = 0; st < 4; ++st)
    #pragma unroll
    for (int j = 0; j < NSUB; ++j)
      #pragma unroll
      for (int r = 0; r < 8; ++r) {
        const int mrow = mbase + st * 16 + (hs ? 8 + r : r);
        C[(size_t)mrow * NDIM + (nt0 + j) * 16 + rl] = acc[st][j][r];
      }
}

// Stage-1 GEMM: grid (2048/256, 16384/64); wave tile 64x32
__global__ __launch_bounds__(256) void wmma_gemm_stage1(const __bf16* __restrict__ Asw,
                                                        const __bf16* __restrict__ Bsw,
                                                        float* __restrict__ C) {
  __shared__ __attribute__((aligned(32))) __bf16 As[2][2048];
  gemm_panel<2>(Asw + (size_t)blockIdx.y * KTILES * 2048, Bsw, C, blockIdx.y * 64, As);
}

// ---------------- stage-1 epilogue: mob_mat_mul nonlinearity, in place ----------------

__global__ __launch_bounds__(256) void mobius_scale_rows(float* __restrict__ Mx,
                                                         const float* __restrict__ xn) {
  __shared__ float red[256];
  const long long base = (long long)blockIdx.x * NDIM;
  float v[8];
  float s = 0.f;
  #pragma unroll
  for (int i = 0; i < 8; ++i) {
    v[i] = Mx[base + threadIdx.x + i * 256];
    s += v[i] * v[i];
  }
  s = block_reduce_sum(s, red);
  const float Mxn = sqrtf(s) + EPSF;
  const float xv  = xn[blockIdx.x];
  const float g   = tanhf((Mxn / xv) * atanh_ref(xv));
  const float prj = fminf(1.0f, MAXNORM / fmaxf(fabsf(g), EPSF));
  const float sc  = g * prj / Mxn;
  #pragma unroll
  for (int i = 0; i < 8; ++i)
    Mx[base + threadIdx.x + i * 256] = v[i] * sc;
}

// ---------------- init: h0 swizzled panel, hn, sync counter ----------------

__global__ __launch_bounds__(256) void init_state(__bf16* __restrict__ Hsw,
                                                  float* __restrict__ hn,
                                                  unsigned* __restrict__ sync) {
  int i = blockIdx.x * blockDim.x + threadIdx.x;
  if (i < BATCH * KDIM) Hsw[i] = (__bf16)EPSF;  // (0 + EPS), layout-invariant constant fill
  if (i < BATCH) hn[i] = EPSF * sqrtf((float)KDIM);
  if (i == 0) *sync = 0u;
}

// ---------------- per-step Mobius epilogue (device fn) ----------------
// h_new = project(mob_add(mob_add(tanh-scale(Wh_row), ux_t), b))
__device__ __forceinline__ void step_epilogue(int b, int t,
                                              const float* __restrict__ Wh,
                                              float* __restrict__ hn,
                                              float* __restrict__ out,
                                              const float* __restrict__ bias,
                                              __bf16* __restrict__ Hsw,
                                              float* red) {
  const int tid = threadIdx.x;
  const long long whb = (long long)b * NDIM;
  const long long ob  = ((long long)b * TSTEPS + t) * NDIM;

  float m[8], ux[8], be[8];
  float s = 0.f;
  #pragma unroll
  for (int i = 0; i < 8; ++i) {
    const int c = tid + i * 256;
    m[i]  = Wh[whb + c];
    ux[i] = out[ob + c] + EPSF;
    be[i] = bias[c] + EPSF;
    s += m[i] * m[i];
  }
  s = block_reduce_sum(s, red);

  const float Mxn  = sqrtf(s) + EPSF;
  const float xv   = hn[b];
  const float g    = tanhf((Mxn / xv) * atanh_ref(xv));
  const float prjm = fminf(1.0f, MAXNORM / fmaxf(fabsf(g), EPSF));
  const float scm  = g * prjm / Mxn;
  const float na   = scm * scm * s;

  float duv = 0.f, nv = 0.f;
  #pragma unroll
  for (int i = 0; i < 8; ++i) {
    m[i] *= scm;
    duv += m[i] * ux[i];
    nv  += ux[i] * ux[i];
  }
  duv = 2.f * block_reduce_sum(duv, red);
  nv  = block_reduce_sum(nv, red);
  float denom = 1.f + duv + na * nv;
  const float ca = (1.f + duv + nv) / denom;
  const float cv = (1.f - na) / denom;

  float a1[8], n1s = 0.f;
  #pragma unroll
  for (int i = 0; i < 8; ++i) { a1[i] = ca * m[i] + cv * ux[i]; n1s += a1[i] * a1[i]; }
  n1s = block_reduce_sum(n1s, red);
  const float n1 = sqrtf(n1s);
  const float p1 = fminf(1.f, MAXNORM / fmaxf(n1, EPSF));
  const float na2 = n1s * p1 * p1;

  float duv2 = 0.f, nv2 = 0.f;
  #pragma unroll
  for (int i = 0; i < 8; ++i) {
    a1[i] *= p1;
    duv2 += a1[i] * be[i];
    nv2  += be[i] * be[i];
  }
  duv2 = 2.f * block_reduce_sum(duv2, red);
  nv2  = block_reduce_sum(nv2, red);
  const float denom2 = 1.f + duv2 + na2 * nv2;
  const float ca2 = (1.f + duv2 + nv2) / denom2;
  const float cv2 = (1.f - na2) / denom2;

  float hv[8], hn2 = 0.f;
  #pragma unroll
  for (int i = 0; i < 8; ++i) { hv[i] = ca2 * a1[i] + cv2 * be[i]; hn2 += hv[i] * hv[i]; }
  hn2 = block_reduce_sum(hn2, red);
  const float p2 = fminf(1.f, MAXNORM / fmaxf(sqrtf(hn2), EPSF));

  float hes = 0.f;
  #pragma unroll
  for (int i = 0; i < 8; ++i) {
    hv[i] *= p2;
    const int c = tid + i * 256;
    out[ob + c] = hv[i];
    // (h+EPS) into swizzled bf16 A-panel (mb = 0) for the next step's GEMM
    const int kt = c >> 5, cc = c & 31;
    const int st = b >> 4, r = b & 15;
    const int ln  = r + (((cc >> 3) & 1) << 4);
    const int pos = (cc & 7) + (((cc >> 4) & 1) << 3);
    Hsw[((size_t)kt * 4 + st) * 512 + ln * 16 + pos] = (__bf16)(hv[i] + EPSF);
    const float he = hv[i] + EPSF;
    hes += he * he;
  }
  hes = block_reduce_sum(hes, red);
  if (tid == 0) hn[b] = sqrtf(hes);
}

// ---------------- persistent recurrent scan: all 256 steps in one kernel ----------------
// 16 resident WGs. Per step: GEMM phase (each WG: 64M x 128N tile of Wh = Hsw @ Wsw),
// grid barrier, epilogue phase (each WG: 4 batch rows), grid barrier.
__global__ __launch_bounds__(256) void recurrent_fused(const __bf16* __restrict__ Wsw,
                                                       __bf16* __restrict__ Hsw,
                                                       float* __restrict__ Wh,
                                                       float* __restrict__ hn,
                                                       float* __restrict__ out,
                                                       const float* __restrict__ bias,
                                                       unsigned* __restrict__ sync) {
  __shared__ __attribute__((aligned(32))) __bf16 As[2][2048];
  __shared__ float red[256];
  const int g = blockIdx.x;   // 0..15
  unsigned target = 0;

  for (int t = 0; t < TSTEPS; ++t) {
    gemm_panel<1>(Hsw, Wsw, Wh, 0, As);       // reads Hsw (h_{t-1}+EPS), writes Wh cols g*128..
    target += 16;
    grid_sync(sync, target);                   // Wh complete, Hsw free to overwrite

    #pragma unroll 1
    for (int rr = 0; rr < 4; ++rr)
      step_epilogue(g * 4 + rr, t, Wh, hn, out, bias, Hsw, red);
    target += 16;
    grid_sync(sync, target);                   // Hsw(t) complete for next GEMM
  }
}

// ---------------- launch ----------------

extern "C" void kernel_launch(void* const* d_in, const int* in_sizes, int n_in,
                              void* d_out, int out_size, void* d_ws, size_t ws_size,
                              hipStream_t stream) {
  (void)in_sizes; (void)n_in; (void)out_size; (void)ws_size;
  const float* x    = (const float*)d_in[0];  // (B,T,I,D) -> 16384 rows x 2048
  const float* w    = (const float*)d_in[1];  // (H,D,H,D) -> 2048x2048
  const float* u    = (const float*)d_in[2];  // (I,D,H,D) -> 2048x2048
  const float* bias = (const float*)d_in[3];  // 2048
  float* out = (float*)d_out;

  char* ws = (char*)d_ws;
  const size_t MB = 1024 * 1024;
  __bf16*  Usw  = (__bf16*)ws;                            // 8 MB  swizzled U
  __bf16*  Wsw  = (__bf16*)(ws + 8 * MB);                 // 8 MB  swizzled W
  __bf16*  Xsw  = (__bf16*)(ws + 16 * MB);                // 64 MB swizzled (x+EPS)
  float*   xn   = (float*)(ws + 80 * MB);                 // 64 KB
  __bf16*  Hsw  = (__bf16*)(ws + 80 * MB + (64u << 10));  // 256 KB swizzled (h+EPS)
  float*   hn   = (float*)(ws + 80 * MB + (320u << 10));  // 4 KB slot
  float*   Wh   = (float*)(ws + 80 * MB + (324u << 10));  // 512 KB
  unsigned* syn = (unsigned*)(ws + 80 * MB + (836u << 10));

  // one-time operand swizzles (weights stay L2-resident afterwards)
  swizzle_weights<<<KTILES * NTILES / 8, 256, 0, stream>>>(u, Usw);
  swizzle_weights<<<KTILES * NTILES / 8, 256, 0, stream>>>(w, Wsw);
  swizzle_x<<<(16384 / 64) * KTILES * 4 / 8, 256, 0, stream>>>(x, Xsw);

  // Stage 1: Ux for all rows (memory order b*T+t), staged in d_out, in-place epilogue
  row_norm<<<16384, 256, 0, stream>>>(x, xn);
  wmma_gemm_stage1<<<dim3(NDIM / 256, 16384 / 64), 256, 0, stream>>>(Xsw, Usw, out);
  mobius_scale_rows<<<16384, 256, 0, stream>>>(out, xn);

  // Recurrent scan: one persistent kernel, 16 WGs, grid-barriered phases
  init_state<<<(BATCH * KDIM + 255) / 256, 256, 0, stream>>>(Hsw, hn, syn);
  recurrent_fused<<<16, 256, 0, stream>>>(Wsw, Hsw, Wh, hn, out, bias, syn);
}
